// LSTMAutoencoder_66159676227725
// MI455X (gfx1250) — matrix operand order repaired and checked by reading, not compiled, not run on hardware
//
#include <hip/hip_runtime.h>

// ---------------------------------------------------------------------------
// LSTM autoencoder, fully fused per 16-row batch tile (one wave32 per tile).
// bf16 WMMA (v_wmma_f32_16x16x32_bf16), fp32 accumulate & cell state.
// Gate activations use gfx1250 native v_tanh_f32 when available.
// ---------------------------------------------------------------------------

typedef __attribute__((ext_vector_type(16))) __bf16 v16bf;
typedef __attribute__((ext_vector_type(8)))  float  v8f;

struct U32B { uint4 lo, hi; };   // 32 bytes = one WMMA A/B operand per lane

#define T_STEPS 30

// weight-tile region offsets (in 1KB tiles); tiles laid out [ntile][kchunk]
#define TW1I 0    // enc1 Wih: K pad 4->32, N=256 : 16 tiles
#define TW1H 16   // enc1 Whh: K=64, N=256        : 32 tiles
#define TW2I 48   // enc2 Wih: K=64, N=128        : 16 tiles
#define TW2H 64   // enc2 Whh: K=32, N=128        :  8 tiles
#define TW3I 72   // dec3 Wih: K=32, N=128        :  8 tiles
#define TW3H 80   // dec3 Whh: K=32, N=128        :  8 tiles
#define TW4I 88   // dec4 Wih: K=32, N=256        : 16 tiles
#define TW4H 104  // dec4 Whh: K=64, N=256        : 32 tiles
#define TWO  136  // W_out : K=64, N pad 4->16    :  2 tiles
#define NTILES 138

__device__ __forceinline__ unsigned short f2bf(float f) {
  unsigned int u = __float_as_uint(f);
  u += 0x7FFFu + ((u >> 16) & 1u);          // round to nearest even
  return (unsigned short)(u >> 16);
}

// gfx1250 has native V_TANH_F32 (TRANS op). sigmoid via tanh: one trans op
// instead of exp2+rcp chains -> ~3x shorter serial elementwise path.
#if __has_builtin(__builtin_amdgcn_tanhf)
__device__ __forceinline__ float fast_tanh(float x) {
  return __builtin_amdgcn_tanhf(x);
}
__device__ __forceinline__ float fast_sig(float x) {
  return __builtin_fmaf(0.5f, __builtin_amdgcn_tanhf(0.5f * x), 0.5f);
}
#else
__device__ __forceinline__ float fast_sig(float x) {
  return __builtin_amdgcn_rcpf(1.0f + __builtin_amdgcn_exp2f(-1.4426950408889634f * x));
}
__device__ __forceinline__ float fast_tanh(float x) {
  return 2.0f * __builtin_amdgcn_rcpf(1.0f + __builtin_amdgcn_exp2f(-2.8853900817779268f * x)) - 1.0f;
}
#endif

// A operand (16x32 bf16) from row-major LDS tile.
// lane 0-15: rows 0-15, K 0-7 & 16-23; lane 16-31: rows 0-15, K 8-15 & 24-31.
__device__ __forceinline__ v16bf load_a(const unsigned short* buf, int strideE,
                                        int kc, int lane) {
  const char* base = (const char*)(buf + (lane & 15) * strideE)
                   + kc * 64 + ((lane < 16) ? 0 : 16);
  U32B p;
  p.lo = *(const uint4*)(base);
  p.hi = *(const uint4*)(base + 32);
  return __builtin_bit_cast(v16bf, p);
}

// B operand (32x16 bf16) from pre-swizzled global tiles (lane-contiguous 32B).
__device__ __forceinline__ v16bf load_b(const uint4* w, int tile, int lane) {
  U32B p;
  p.lo = w[tile * 64 + lane * 2];
  p.hi = w[tile * 64 + lane * 2 + 1];
  return __builtin_bit_cast(v16bf, p);
}

// One LSTM step for a 16-row batch tile.
// inBuf/hBuf: row-major bf16 LDS tiles. wih/whh: swizzled B tiles [nt][kc].
// bb: per-lane bias (column = nt*16 + lane%16). c: cell state in D layout.
template <int H, int KCI, int KCH>
__device__ __forceinline__ void lstm_step(const unsigned short* inBuf, int inStride,
                                          unsigned short* hBuf, int hStride,
                                          const uint4* wih, const uint4* whh,
                                          const float (&bb)[H / 4],
                                          float (&c)[H / 16][8], int lane) {
  constexpr int HB = H / 16;
  v16bf Ain[KCI], Ah[KCH];
#pragma unroll
  for (int k = 0; k < KCI; k++) Ain[k] = load_a(inBuf, inStride, k, lane);
#pragma unroll
  for (int k = 0; k < KCH; k++) Ah[k] = load_a(hBuf, hStride, k, lane);

#pragma unroll
  for (int j = 0; j < HB; j++) {
    v8f z[4];
#pragma unroll
    for (int g = 0; g < 4; g++) {
      const int nt = g * HB + j;
      const float b = bb[nt];
      v8f acc = {b, b, b, b, b, b, b, b};
#pragma unroll
      for (int k = 0; k < KCI; k++)
        acc = __builtin_amdgcn_wmma_f32_16x16x32_bf16(
            false, Ain[k], false, load_b(wih, nt * KCI + k, lane),
            (short)0, acc, false, false);
#pragma unroll
      for (int k = 0; k < KCH; k++)
        acc = __builtin_amdgcn_wmma_f32_16x16x32_bf16(
            false, Ah[k], false, load_b(whh, nt * KCH + k, lane),
            (short)0, acc, false, false);
      z[g] = acc;
    }
    const int row0 = (lane < 16) ? 0 : 8;
    const int col = j * 16 + (lane & 15);
#pragma unroll
    for (int e = 0; e < 8; e++) {
      const float iv = fast_sig(z[0][e]);
      const float fv = fast_sig(z[1][e]);
      const float gv = fast_tanh(z[2][e]);
      const float ov = fast_sig(z[3][e]);
      const float cn = fv * c[j][e] + iv * gv;
      c[j][e] = cn;
      hBuf[(row0 + e) * hStride + col] = f2bf(ov * fast_tanh(cn));
    }
  }
}

__device__ __forceinline__ void zero_lds(unsigned short* p, int dwords, int lane) {
  unsigned int* q = (unsigned int*)p;
  for (int i = lane; i < dwords; i += 32) q[i] = 0u;
}

// ---------------------------------------------------------------------------
// Main fused kernel: 128 threads = 4 waves, each wave owns 16 batch rows.
// ---------------------------------------------------------------------------
__global__ __launch_bounds__(128, 1) void lstm_main(
    const float* __restrict__ x, const uint4* __restrict__ wt,
    const float* __restrict__ bias, float* __restrict__ out) {
  // per-wave LDS staging (row strides padded for conflict-free b128 reads)
  // x:16x40u16, h1:16x72, h2:16x40, h3:16x40, h4:16x72  => 4224 u16 / wave
  __shared__ unsigned short lds[4 * 4224];
  const int lane = threadIdx.x & 31;
  const int wave = threadIdx.x >> 5;
  const int batchBase = (blockIdx.x * 4 + wave) * 16;

  unsigned short* xbuf = lds + wave * 4224;
  unsigned short* h1 = xbuf + 640;
  unsigned short* h2 = h1 + 1152;
  unsigned short* h3 = h2 + 640;
  unsigned short* h4 = h3 + 640;

  const uint4* W1I = wt + TW1I * 64;
  const uint4* W1H = wt + TW1H * 64;
  const uint4* W2I = wt + TW2I * 64;
  const uint4* W2H = wt + TW2H * 64;
  const uint4* W3I = wt + TW3I * 64;
  const uint4* W3H = wt + TW3H * 64;
  const uint4* W4I = wt + TW4I * 64;
  const uint4* W4H = wt + TW4H * 64;
  const uint4* WO  = wt + TWO * 64;

  // per-lane bias preload (column = nt*16 + lane%16)
  const int lc = lane & 15;
  float b1[16], b2[8], b3[8], b4[16];
#pragma unroll
  for (int nt = 0; nt < 16; nt++) b1[nt] = bias[nt * 16 + lc];
#pragma unroll
  for (int nt = 0; nt < 8; nt++) b2[nt] = bias[256 + nt * 16 + lc];
#pragma unroll
  for (int nt = 0; nt < 8; nt++) b3[nt] = bias[384 + nt * 16 + lc];
#pragma unroll
  for (int nt = 0; nt < 16; nt++) b4[nt] = bias[512 + nt * 16 + lc];
  const float bo = (lc < 4) ? bias[768 + lc] : 0.0f;

  zero_lds(xbuf, 2112, lane);  // zeros x padding + h0 states

  float c1[4][8] = {}, c2[2][8] = {};

  // ---------------- Phase A: enc1 -> enc2, interleaved per timestep --------
#pragma unroll 1
  for (int t = 0; t < T_STEPS; t++) {
    asm volatile("" ::: "memory");  // keep weight-tile loads inside the loop
    if (lane < 16) {                // stage x[b, t, 0:4] as bf16 (cols 4..31 stay 0)
      const float4 xv = *(const float4*)(x + ((size_t)(batchBase + lane) * T_STEPS + t) * 4);
      unsigned int* r = (unsigned int*)(xbuf + lane * 40);
      r[0] = (unsigned int)f2bf(xv.x) | ((unsigned int)f2bf(xv.y) << 16);
      r[1] = (unsigned int)f2bf(xv.z) | ((unsigned int)f2bf(xv.w) << 16);
    }
    lstm_step<64, 1, 2>(xbuf, 40, h1, 72, W1I, W1H, b1, c1, lane);  // enc1
    lstm_step<32, 2, 1>(h1, 72, h2, 40, W2I, W2H, b2, c2, lane);    // enc2
  }
  // h2 now holds the bottleneck (constant decoder input)

  // ---------------- Phase B: dec3 -> dec4 -> output projection -------------
  float c3[2][8] = {}, c4[4][8] = {};
#pragma unroll 1
  for (int t = 0; t < T_STEPS; t++) {
    asm volatile("" ::: "memory");
    lstm_step<32, 1, 1>(h2, 40, h3, 40, W3I, W3H, b3, c3, lane);    // dec3
    lstm_step<64, 1, 2>(h3, 40, h4, 72, W4I, W4H, b4, c4, lane);    // dec4

    // out = h4 @ W_out.T + b_out  (N padded to 16, only cols 0..3 valid)
    v8f acc = {bo, bo, bo, bo, bo, bo, bo, bo};
    acc = __builtin_amdgcn_wmma_f32_16x16x32_bf16(
        false, load_a(h4, 72, 0, lane), false, load_b(WO, 0, lane),
        (short)0, acc, false, false);
    acc = __builtin_amdgcn_wmma_f32_16x16x32_bf16(
        false, load_a(h4, 72, 1, lane), false, load_b(WO, 1, lane),
        (short)0, acc, false, false);
    if (lc < 4) {
      const int row0 = (lane < 16) ? 0 : 8;
#pragma unroll
      for (int e = 0; e < 8; e++)
        out[((size_t)(batchBase + row0 + e) * T_STEPS + t) * 4 + lc] = acc[e];
    }
  }
}

// ---------------------------------------------------------------------------
// Prep: swizzle fp32 weights (4H x in, row-major) into bf16 WMMA B tiles.
// B[k][n] = W[n][k]; tile (nt,kc): lane holds column n=nt*16+lane%16,
// K = kc*32 + (lane<16?0:8) + (j<4?0:16) + 2*(j&3) (+1 in high half) for dword j=0..7.
// ---------------------------------------------------------------------------
__global__ void prep_weights(const float* __restrict__ W, unsigned int* __restrict__ dst,
                             int Nrows, int Kcols, int KC, int NT) {
  const int total = NT * KC * 256;
  for (int i = blockIdx.x * blockDim.x + threadIdx.x; i < total;
       i += gridDim.x * blockDim.x) {
    const int tile = i >> 8, w = i & 255;
    const int lane = w >> 3, j = w & 7;
    const int nt = tile / KC, kc = tile - nt * KC;
    const int n = nt * 16 + (lane & 15);
    const int k0 = kc * 32 + ((lane >> 4) << 3) + ((j & 4) << 2) + 2 * (j & 3);
    const float v0 = (n < Nrows && k0 < Kcols) ? W[n * Kcols + k0] : 0.0f;
    const float v1 = (n < Nrows && (k0 + 1) < Kcols) ? W[n * Kcols + k0 + 1] : 0.0f;
    dst[i] = (unsigned int)f2bf(v0) | ((unsigned int)f2bf(v1) << 16);
  }
}

__global__ void prep_bias(const float* b1i, const float* b1h, const float* b2i,
                          const float* b2h, const float* b3i, const float* b3h,
                          const float* b4i, const float* b4h, const float* bo,
                          float* __restrict__ dst) {
  const int i = blockIdx.x * blockDim.x + threadIdx.x;
  if (i < 256)       dst[i] = b1i[i] + b1h[i];
  else if (i < 384)  dst[i] = b2i[i - 256] + b2h[i - 256];
  else if (i < 512)  dst[i] = b3i[i - 384] + b3h[i - 384];
  else if (i < 768)  dst[i] = b4i[i - 512] + b4h[i - 512];
  else if (i < 772)  dst[i] = bo[i - 768];
}

extern "C" void kernel_launch(void* const* d_in, const int* in_sizes, int n_in,
                              void* d_out, int out_size, void* d_ws, size_t ws_size,
                              hipStream_t stream) {
  (void)in_sizes; (void)n_in; (void)out_size; (void)ws_size;
  const float* x = (const float*)d_in[0];
  unsigned int* wt32 = (unsigned int*)d_ws;
  float* biases = (float*)((char*)d_ws + NTILES * 1024);

  auto prep = [&](const float* W, int tileOfs, int Nrows, int Kcols, int KC, int NT) {
    const int total = NT * KC * 256;
    prep_weights<<<(total + 255) / 256, 256, 0, stream>>>(
        W, wt32 + tileOfs * 256, Nrows, Kcols, KC, NT);
  };
  prep((const float*)d_in[1],  TW1I, 256, 4, 1, 16);   // enc1 Wih
  prep((const float*)d_in[2],  TW1H, 256, 64, 2, 16);  // enc1 Whh
  prep((const float*)d_in[5],  TW2I, 128, 64, 2, 8);   // enc2 Wih
  prep((const float*)d_in[6],  TW2H, 128, 32, 1, 8);   // enc2 Whh
  prep((const float*)d_in[9],  TW3I, 128, 32, 1, 8);   // dec3 Wih
  prep((const float*)d_in[10], TW3H, 128, 32, 1, 8);   // dec3 Whh
  prep((const float*)d_in[13], TW4I, 256, 32, 1, 16);  // dec4 Wih
  prep((const float*)d_in[14], TW4H, 256, 64, 2, 16);  // dec4 Whh
  prep((const float*)d_in[17], TWO,  4, 64, 2, 1);     // W_out

  prep_bias<<<4, 256, 0, stream>>>(
      (const float*)d_in[3], (const float*)d_in[4],
      (const float*)d_in[7], (const float*)d_in[8],
      (const float*)d_in[11], (const float*)d_in[12],
      (const float*)d_in[15], (const float*)d_in[16],
      (const float*)d_in[18], biases);

  // 16384 batch rows / (4 waves * 16 rows) = 256 blocks
  lstm_main<<<256, 128, 0, stream>>>(x, (const uint4*)d_ws, biases, (float*)d_out);
}
